// FreModule_1305670058104
// MI455X (gfx1250) — matrix operand order repaired and checked by reading, not compiled, not run on hardware
//
#include <hip/hip_runtime.h>
#include <hip/hip_bf16.h>
#include <math.h>

// ---------------------------------------------------------------------------
// FreModule pipeline for MI455X (gfx1250, wave32).
// All channel-mixing GEMMs (1x1 convs, kv conv, attention-apply+proj, refine
// proj, final blend) go through v_wmma_f32_16x16x32_f16 (K=64 -> 2 WMMAs per
// 16x16 tile, 2 pixel-tiles per wave). FFT path replaced by exact
// 4-coefficient DFT (thresholds are provably in {0,1}).
// Epilogues are compile-time template variants -> no branches in store loop.
// ---------------------------------------------------------------------------

typedef __attribute__((ext_vector_type(16))) _Float16 v16h;
typedef __attribute__((ext_vector_type(8)))  float    v8f;

#define NP 65536          // 256*256 pixels
#define BATCH 8
#define DIMC 64
#define TWO_PI_OVER_256 0.0245436926061702596f

// ---------------------------------------------------------------------------
// WMMA GEMM: Y[b, o, n] = sum_c W[(b,)o,c] * X[b, c, n]   (Cin == 64)
// X base pointer + CinStride gives a channel-window view (for kv "v" slice).
// Template flags: PERB  = per-batch weights (W has (B,Cout,64))
//                 BIAS  = add bias[o]
//                 EPI   = out = acc*scale[o] + resid[b,o,n]*rscale[o]
// grid = (N/256, Cout/16, B), block = 256 (8 waves); each wave computes two
// 16x16 output tiles (px and px+16) reusing one 16x64 A fragment (4 WMMAs).
// ---------------------------------------------------------------------------
template <bool PERB, bool BIAS, bool EPI>
__global__ void gemm64_wmma(const float* __restrict__ X, int CinStride,
                            const float* __restrict__ W,
                            float* __restrict__ Y, int Cout, int N,
                            const float* __restrict__ bias,
                            const float* __restrict__ scale,
                            const float* __restrict__ resid,
                            const float* __restrict__ rscale)
{
    const int lane = threadIdx.x & 31;
    const int wave = threadIdx.x >> 5;
    const int b    = blockIdx.z;
    const int row0 = blockIdx.y * 16;
    const int hi   = lane >> 4;          // half-wave select
    const int M    = lane & 15;          // A row within tile
    const size_t px0 = (size_t)blockIdx.x * 256 + wave * 32 + (lane & 15);
    const size_t px1 = px0 + 16;

    const float* Wb = PERB ? (W + (size_t)b * Cout * 64) : W;
    const float* Xb = X + (size_t)b * CinStride * N;

    // ---- A (weights) 16x64 as two 16x32 f16 fragments (contiguous K, matching B) ----
    v16h a0, a1;
#pragma unroll
    for (int i = 0; i < 16; ++i) {
        int j = i >> 1, p = i & 1;
        int k = hi * 16 + i;   // K in [0,32)
        a0[i] = (_Float16)Wb[(row0 + M) * 64 + k];
        a1[i] = (_Float16)Wb[(row0 + M) * 64 + 32 + k];
    }
    // ---- B (activations) 64x16 as two 32x16 f16 fragments per pixel tile ---
    // lane: N-col = lane&15, K rows = hi*16 + i
    v16h b0, b1, b2, b3;
    const int kb = hi * 16;
#pragma unroll
    for (int i = 0; i < 16; ++i) {
        b0[i] = (_Float16)Xb[(size_t)(kb + i) * N + px0];
        b1[i] = (_Float16)Xb[(size_t)(32 + kb + i) * N + px0];
        b2[i] = (_Float16)Xb[(size_t)(kb + i) * N + px1];
        b3[i] = (_Float16)Xb[(size_t)(32 + kb + i) * N + px1];
    }

    v8f c0 = {}, c1 = {};
    c0 = __builtin_amdgcn_wmma_f32_16x16x32_f16(false, a0, false, b0,
                                                (short)0, c0, false, false);
    c1 = __builtin_amdgcn_wmma_f32_16x16x32_f16(false, a0, false, b2,
                                                (short)0, c1, false, false);
    c0 = __builtin_amdgcn_wmma_f32_16x16x32_f16(false, a1, false, b1,
                                                (short)0, c0, false, false);
    c1 = __builtin_amdgcn_wmma_f32_16x16x32_f16(false, a1, false, b3,
                                                (short)0, c1, false, false);

    // ---- store (C/D layout: VGPR r -> row r + hi*8, col = lane&15) ----
#pragma unroll
    for (int r = 0; r < 8; ++r) {
        int oc = row0 + r + hi * 8;
        size_t base = ((size_t)b * Cout + oc) * N;
        float v0 = c0[r], v1 = c1[r];
        if constexpr (BIAS) { float bb = bias[oc]; v0 += bb; v1 += bb; }
        if constexpr (EPI) {
            float sc = scale[oc], rs = rscale[oc];
            v0 = v0 * sc + resid[base + px0] * rs;
            v1 = v1 * sc + resid[base + px1] * rs;
        }
        Y[base + px0] = v0;
        Y[base + px1] = v1;
    }
}

// ---------------------------------------------------------------------------
// Bilinear resize (8,3,128,128) -> (8,3,256,256), half-pixel centers, clamp.
// ---------------------------------------------------------------------------
__global__ void resize_k(const float* __restrict__ in, float* __restrict__ out)
{
    int idx = blockIdx.x * 256 + threadIdx.x;        // 8*3*65536 threads
    int p  = idx & (NP - 1);
    int bc = idx >> 16;
    int oy = p >> 8, ox = p & 255;
    float fy = 0.5f * oy - 0.25f;
    float fx = 0.5f * ox - 0.25f;
    int y0 = (int)floorf(fy); float wy = fy - y0;
    int x0 = (int)floorf(fx); float wx = fx - x0;
    int y0c = min(127, max(0, y0)),   y1c = min(127, max(0, y0 + 1));
    int x0c = min(127, max(0, x0)),   x1c = min(127, max(0, x0 + 1));
    const float* pi = in + (size_t)bc * 128 * 128;
    float v = (1.f - wy) * ((1.f - wx) * pi[y0c * 128 + x0c] + wx * pi[y0c * 128 + x1c])
            +        wy  * ((1.f - wx) * pi[y1c * 128 + x0c] + wx * pi[y1c * 128 + x1c]);
    out[idx] = v;
}

// ---------------------------------------------------------------------------
// 3x3 conv, 3 -> 64 channels, pad 1. grid=(256,64,8)
// ---------------------------------------------------------------------------
__global__ void conv3x3_k(const float* __restrict__ in,
                          const float* __restrict__ w, float* __restrict__ out)
{
    int p = blockIdx.x * 256 + threadIdx.x;
    int o = blockIdx.y, b = blockIdx.z;
    int y = p >> 8, x = p & 255;
    float acc = 0.f;
#pragma unroll
    for (int ci = 0; ci < 3; ++ci) {
        const float* pi = in + (((size_t)b * 3 + ci) << 16);
        const float* pw = w + (o * 3 + ci) * 9;
#pragma unroll
        for (int ky = 0; ky < 3; ++ky) {
            int yy = y + ky - 1;
            if ((unsigned)yy > 255u) continue;
#pragma unroll
            for (int kx = 0; kx < 3; ++kx) {
                int xx = x + kx - 1;
                if ((unsigned)xx > 255u) continue;
                acc += pw[ky * 3 + kx] * pi[(yy << 8) + xx];
            }
        }
    }
    out[(((size_t)b * DIMC + o) << 16) + p] = acc;
}

// ---------------------------------------------------------------------------
// Depthwise 3x3 conv, pad 1. grid=(256,C,8)
// ---------------------------------------------------------------------------
__global__ void dwconv3x3_k(const float* __restrict__ in,
                            const float* __restrict__ w,
                            float* __restrict__ out, int C)
{
    int p = blockIdx.x * 256 + threadIdx.x;
    int c = blockIdx.y, b = blockIdx.z;
    int y = p >> 8, x = p & 255;
    const float* pi = in + (((size_t)b * C + c) << 16);
    const float* pw = w + c * 9;
    float acc = 0.f;
#pragma unroll
    for (int ky = 0; ky < 3; ++ky) {
        int yy = y + ky - 1;
        if ((unsigned)yy > 255u) continue;
#pragma unroll
        for (int kx = 0; kx < 3; ++kx) {
            int xx = x + kx - 1;
            if ((unsigned)xx > 255u) continue;
            acc += pw[ky * 3 + kx] * pi[(yy << 8) + xx];
        }
    }
    out[(((size_t)b * C + c) << 16) + p] = acc;
}

// ---------------------------------------------------------------------------
// Per-(b,c) mean over pixels. grid = B*64 blocks. Cstride = channels/batch.
// ---------------------------------------------------------------------------
__global__ void mean_k(const float* __restrict__ X, float* __restrict__ out,
                       int Cstride, int N)
{
    int b = blockIdx.x >> 6, c = blockIdx.x & 63, t = threadIdx.x;
    const float* p = X + ((size_t)b * Cstride + c) * N;
    float s = 0.f;
    for (int n = t; n < N; n += 256) s += p[n];
    __shared__ float red[256];
    red[t] = s; __syncthreads();
    for (int st = 128; st > 0; st >>= 1) {
        if (t < st) red[t] += red[t + st];
        __syncthreads();
    }
    if (t == 0) out[blockIdx.x] = red[0] / (float)N;
}

__global__ void sumsq_k(const float* __restrict__ X, float* __restrict__ out,
                        int Cstride, int N)
{
    int b = blockIdx.x >> 6, c = blockIdx.x & 63, t = threadIdx.x;
    const float* p = X + ((size_t)b * Cstride + c) * N;
    float s = 0.f;
    for (int n = t; n < N; n += 256) { float v = p[n]; s += v * v; }
    __shared__ float red[256];
    red[t] = s; __syncthreads();
    for (int st = 128; st > 0; st >>= 1) {
        if (t < st) red[t] += red[t + st];
        __syncthreads();
    }
    if (t == 0) out[blockIdx.x] = red[0];
}

__global__ void avgmax_k(const float* __restrict__ X, float* __restrict__ avg,
                         float* __restrict__ mx, int Cstride, int N)
{
    int b = blockIdx.x >> 6, c = blockIdx.x & 63, t = threadIdx.x;
    const float* p = X + ((size_t)b * Cstride + c) * N;
    float s = 0.f, m = -1e30f;
    for (int n = t; n < N; n += 256) { float v = p[n]; s += v; m = fmaxf(m, v); }
    __shared__ float rs[256], rm[256];
    rs[t] = s; rm[t] = m; __syncthreads();
    for (int st = 128; st > 0; st >>= 1) {
        if (t < st) { rs[t] += rs[t + st]; rm[t] = fmaxf(rm[t], rm[t + st]); }
        __syncthreads();
    }
    if (t == 0) { avg[blockIdx.x] = rs[0] / (float)N; mx[blockIdx.x] = rm[0]; }
}

// ---------------------------------------------------------------------------
// Threshold MLP: pooled(8,64) -> gelu(@rate_w1^T) -> @rate_w2^T -> sigmoid
// h_ = int(2*thr0), w_ = int(2*thr1)  (each in {0,1}).  <<<1, 8>>>
// ---------------------------------------------------------------------------
__global__ void thr_k(const float* __restrict__ pooled,
                      const float* __restrict__ w1,   // (8,64)
                      const float* __restrict__ w2,   // (2,8)
                      int* __restrict__ thr)
{
    int b = threadIdx.x;
    if (b >= BATCH) return;
    float g[8];
#pragma unroll
    for (int j = 0; j < 8; ++j) {
        float s = 0.f;
        for (int c = 0; c < 64; ++c) s += pooled[b * 64 + c] * w1[j * 64 + c];
        g[j] = 0.5f * s * (1.f + erff(s * 0.70710678118654752f)); // exact gelu
    }
    float t0 = 0.f, t1 = 0.f;
#pragma unroll
    for (int j = 0; j < 8; ++j) { t0 += g[j] * w2[j]; t1 += g[j] * w2[8 + j]; }
    float th0 = 1.f / (1.f + expf(-t0));
    float th1 = 1.f / (1.f + expf(-t1));
    thr[b * 2 + 0] = (int)(2.f * th0);
    thr[b * 2 + 1] = (int)(2.f * th1);
}

// ---------------------------------------------------------------------------
// 4-coefficient DFT per (b,c) image: F[k] for k in {0,-1}^2.
// F stored as [F00, F0m.re, F0m.im, Fm0.re, Fm0.im, Fmm.re, Fmm.im, pad].
// grid = 512 blocks (b*64+c), block = 256.
// ---------------------------------------------------------------------------
__global__ void dft4_k(const float* __restrict__ X, float* __restrict__ F)
{
    int bc = blockIdx.x, t = threadIdx.x;
    __shared__ float ct[256], st[256];
    float ang = TWO_PI_OVER_256 * (float)t;
    __sincosf(ang, &st[t], &ct[t]);
    __syncthreads();
    const float* p = X + ((size_t)bc << 16);
    float f00 = 0, f0mr = 0, f0mi = 0, fm0r = 0, fm0i = 0, fmmr = 0, fmmi = 0;
    for (int n = t; n < NP; n += 256) {
        int n1 = n >> 8, n2 = n & 255;
        float v = p[n];
        float c1 = ct[n1], s1 = st[n1], c2 = ct[n2], s2 = st[n2];
        f00 += v;
        f0mr += v * c2; f0mi += v * s2;      // k2=-1: e^{+i a2}
        fm0r += v * c1; fm0i += v * s1;
        float c12 = c1 * c2 - s1 * s2, s12 = s1 * c2 + c1 * s2;
        fmmr += v * c12; fmmi += v * s12;
    }
    __shared__ float red[256];
    float vals[7] = { f00, f0mr, f0mi, fm0r, fm0i, fmmr, fmmi };
    for (int e = 0; e < 7; ++e) {
        red[t] = vals[e]; __syncthreads();
        for (int s = 128; s > 0; s >>= 1) {
            if (t < s) red[t] += red[t + s];
            __syncthreads();
        }
        if (t == 0) F[bc * 8 + e] = red[0];
        __syncthreads();
    }
}

// ---------------------------------------------------------------------------
// Reconstruct low/high:
//   mask nonempty (h_==1 && w_==1):
//     low_c = (F00 + F0m*E2 + Fm0*E1 + Fmm*E1*E2)/65536,  E = e^{-2*pi*i n/256}
//     low = |low_c|, high = |x - low_c|
//   else: low = 0, high = |x|
// grid=(256,64,8)
// ---------------------------------------------------------------------------
__global__ void lowhigh_k(const float* __restrict__ X,
                          const float* __restrict__ F,
                          const int* __restrict__ thr,
                          float* __restrict__ high, float* __restrict__ low)
{
    int p = blockIdx.x * 256 + threadIdx.x;
    int c = blockIdx.y, b = blockIdx.z;
    size_t i = (((size_t)b * DIMC + c) << 16) + p;
    float xv = X[i];
    float lo, hi;
    if (thr[b * 2] == 1 && thr[b * 2 + 1] == 1) {
        const float* f = F + (b * DIMC + c) * 8;
        int n1 = p >> 8, n2 = p & 255;
        float s1, c1, s2, c2;
        __sincosf(TWO_PI_OVER_256 * (float)n1, &s1, &c1);
        __sincosf(TWO_PI_OVER_256 * (float)n2, &s2, &c2);
        float c12 = c1 * c2 - s1 * s2, s12 = s1 * c2 + c1 * s2;
        // multiply by conj twiddles (E = cos - i sin)
        float re = f[0]
                 + (f[1] * c2 + f[2] * s2)
                 + (f[3] * c1 + f[4] * s1)
                 + (f[5] * c12 + f[6] * s12);
        float im = (f[2] * c2 - f[1] * s2)
                 + (f[4] * c1 - f[3] * s1)
                 + (f[6] * c12 - f[5] * s12);
        re *= (1.0f / 65536.0f);
        im *= (1.0f / 65536.0f);
        lo = sqrtf(re * re + im * im);
        float dr = xv - re;
        hi = sqrtf(dr * dr + im * im);
    } else {
        lo = 0.f;
        hi = fabsf(xv);
    }
    low[i] = lo;
    high[i] = hi;
}

// ---------------------------------------------------------------------------
// Per-(b,head) 8x8 Gram of q,k over N, fold in l2 norms + temp, softmax.
// grid = 64 blocks (b*8+h), block = 256.  kv layout (B,128,N): k = ch 0..63.
// ---------------------------------------------------------------------------
__global__ void gram_k(const float* __restrict__ q,
                       const float* __restrict__ kv,
                       const float* __restrict__ qss,
                       const float* __restrict__ kss,
                       const float* __restrict__ temp,
                       float* __restrict__ attn, int N)
{
    int b = blockIdx.x >> 3, h = blockIdx.x & 7, t = threadIdx.x;
    const float* qb = q  + ((size_t)b * 64  + h * 8) * N;
    const float* kb = kv + ((size_t)b * 128 + h * 8) * N;
    float acc[8][8];
#pragma unroll
    for (int cc = 0; cc < 8; ++cc)
#pragma unroll
        for (int d = 0; d < 8; ++d) acc[cc][d] = 0.f;
    for (int n = t; n < N; n += 256) {
        float qv[8], kk[8];
#pragma unroll
        for (int cc = 0; cc < 8; ++cc) {
            qv[cc] = qb[(size_t)cc * N + n];
            kk[cc] = kb[(size_t)cc * N + n];
        }
#pragma unroll
        for (int cc = 0; cc < 8; ++cc)
#pragma unroll
            for (int d = 0; d < 8; ++d) acc[cc][d] += qv[cc] * kk[d];
    }
    __shared__ float red[256];
    __shared__ float g[64];
    for (int e = 0; e < 64; ++e) {
        red[t] = acc[e >> 3][e & 7]; __syncthreads();
        for (int s = 128; s > 0; s >>= 1) {
            if (t < s) red[t] += red[t + s];
            __syncthreads();
        }
        if (t == 0) g[e] = red[0];
        __syncthreads();
    }
    if (t < 8) {
        int cc = t;
        float qn = fmaxf(sqrtf(qss[b * 64 + h * 8 + cc]), 1e-12f);
        float row[8];
        float mxv = -1e30f;
#pragma unroll
        for (int d = 0; d < 8; ++d) {
            float kn = fmaxf(sqrtf(kss[b * 64 + h * 8 + d]), 1e-12f);
            float v = g[cc * 8 + d] / (qn * kn) * temp[h];
            row[d] = v; mxv = fmaxf(mxv, v);
        }
        float s = 0.f;
#pragma unroll
        for (int d = 0; d < 8; ++d) { row[d] = expf(row[d] - mxv); s += row[d]; }
#pragma unroll
        for (int d = 0; d < 8; ++d)
            attn[((b * 8 + h) * 8 + cc) * 8 + d] = row[d] / s;
    }
}

// ---------------------------------------------------------------------------
// W_eff[b][o][dg] = sum_{cl<8} projw[o, h*8+cl] * attn[b,h,cl,dl]
// (proj folded with block-diagonal channel attention).  <<<8, 256>>>
// ---------------------------------------------------------------------------
__global__ void weff_k(const float* __restrict__ attn,
                       const float* __restrict__ projw,
                       float* __restrict__ weff)
{
    int b = blockIdx.x;
    for (int e = threadIdx.x; e < 4096; e += 256) {
        int o = e >> 6, dg = e & 63;
        int h = dg >> 3, dl = dg & 7;
        float s = 0.f;
#pragma unroll
        for (int cl = 0; cl < 8; ++cl)
            s += projw[o * 64 + h * 8 + cl] * attn[((b * 8 + h) * 8 + cl) * 8 + dl];
        weff[((size_t)b * 64 + o) * 64 + dg] = s;
    }
}

// ---------------------------------------------------------------------------
// Channel max + mean of high2 -> (B,2,N). grid=(256,1,8)
// ---------------------------------------------------------------------------
__global__ void chan_maxmean_k(const float* __restrict__ X,
                               float* __restrict__ mxmn)
{
    int p = blockIdx.x * 256 + threadIdx.x;
    int b = blockIdx.z;
    float m = -1e30f, s = 0.f;
    for (int c = 0; c < 64; ++c) {
        float v = X[(((size_t)b * 64 + c) << 16) + p];
        m = fmaxf(m, v); s += v;
    }
    mxmn[(((size_t)b * 2 + 0) << 16) + p] = m;
    mxmn[(((size_t)b * 2 + 1) << 16) + p] = s * (1.0f / 64.0f);
}

// ---------------------------------------------------------------------------
// 7x7 conv 2->1 + sigmoid. grid=(256,1,8)
// ---------------------------------------------------------------------------
__global__ void conv7x7_sig_k(const float* __restrict__ in,
                              const float* __restrict__ w,  // (2,7,7)
                              float* __restrict__ sw)
{
    int p = blockIdx.x * 256 + threadIdx.x;
    int b = blockIdx.z;
    int y = p >> 8, x = p & 255;
    float acc = 0.f;
    for (int ci = 0; ci < 2; ++ci) {
        const float* pi = in + (((size_t)b * 2 + ci) << 16);
        const float* pw = w + ci * 49;
        for (int ky = 0; ky < 7; ++ky) {
            int yy = y + ky - 3;
            if ((unsigned)yy > 255u) continue;
            for (int kx = 0; kx < 7; ++kx) {
                int xx = x + kx - 3;
                if ((unsigned)xx > 255u) continue;
                acc += pw[ky * 7 + kx] * pi[(yy << 8) + xx];
            }
        }
    }
    sw[((size_t)b << 16) + p] = 1.f / (1.f + expf(-acc));
}

// ---------------------------------------------------------------------------
// Channel gate: cw[b,c] = sigmoid( (relu(avg@W1)+relu(max@W1)) @ W2 )
// <<<8, 64>>>
// ---------------------------------------------------------------------------
__global__ void cw_k(const float* __restrict__ avg, const float* __restrict__ mx,
                     const float* __restrict__ w1,  // (4,64)
                     const float* __restrict__ w2,  // (64,4)
                     float* __restrict__ cw)
{
    int b = blockIdx.x, c = threadIdx.x;
    float ha[4], hm[4];
#pragma unroll
    for (int j = 0; j < 4; ++j) {
        float sa = 0.f, sm = 0.f;
        for (int cc = 0; cc < 64; ++cc) {
            sa += avg[b * 64 + cc] * w1[j * 64 + cc];
            sm += mx[b * 64 + cc] * w1[j * 64 + cc];
        }
        ha[j] = fmaxf(sa, 0.f); hm[j] = fmaxf(sm, 0.f);
    }
    float s = 0.f;
#pragma unroll
    for (int j = 0; j < 4; ++j) s += (ha[j] + hm[j]) * w2[c * 4 + j];
    cw[b * 64 + c] = 1.f / (1.f + expf(-s));
}

// ---------------------------------------------------------------------------
// out = low2*sw[b,p] + high2*cw[b,c].  grid=(256,64,8)
// ---------------------------------------------------------------------------
__global__ void combine_k(const float* __restrict__ low2,
                          const float* __restrict__ high2,
                          const float* __restrict__ sw,
                          const float* __restrict__ cw,
                          float* __restrict__ out)
{
    int p = blockIdx.x * 256 + threadIdx.x;
    int c = blockIdx.y, b = blockIdx.z;
    size_t i = (((size_t)b * 64 + c) << 16) + p;
    out[i] = low2[i] * sw[((size_t)b << 16) + p] + high2[i] * cw[b * 64 + c];
}

// ===========================================================================
extern "C" void kernel_launch(void* const* d_in, const int* in_sizes, int n_in,
                              void* d_out, int out_size, void* d_ws, size_t ws_size,
                              hipStream_t stream)
{
    (void)in_sizes; (void)n_in; (void)out_size; (void)ws_size;

    const float* x        = (const float*)d_in[0];   // (8,3,128,128)
    const float* y        = (const float*)d_in[1];   // (8,64,256,256)
    const float* conv1_w  = (const float*)d_in[2];   // (64,3,3,3)
    const float* rate_w1  = (const float*)d_in[3];   // (8,64)
    const float* rate_w2  = (const float*)d_in[4];   // (2,8)
    const float* cca_temp = (const float*)d_in[5];   // (3,8)
    const float* cca_qw   = (const float*)d_in[6];   // (3,64,64)
    const float* cca_qdw  = (const float*)d_in[7];   // (3,64,9)
    const float* cca_kvw  = (const float*)d_in[8];   // (3,128,64)
    const float* cca_kvdw = (const float*)d_in[9];   // (3,128,9)
    const float* cca_projw= (const float*)d_in[10];  // (3,64,64)
    const float* spat_w   = (const float*)d_in[11];  // (2,7,7)
    const float* cg_w1    = (const float*)d_in[12];  // (4,64)
    const float* cg_w2    = (const float*)d_in[13];  // (64,4)
    const float* rproj_w  = (const float*)d_in[14];  // (64,64)
    const float* rproj_b  = (const float*)d_in[15];  // (64)
    const float* para1    = (const float*)d_in[16];  // (64)
    const float* para2    = (const float*)d_in[17];  // (64)
    float* out = (float*)d_out;

    // ---- workspace arena (floats) ----
    const size_t IMG  = (size_t)BATCH * DIMC * NP;   // 33.55M floats
    const size_t IMG2 = 2 * IMG;
    float* ws = (float*)d_ws;
    size_t off = 0;
    auto alloc = [&](size_t n) { float* p = ws + off; off += n; return p; };
    float* XR   = alloc((size_t)BATCH * 3 * NP);   // resized x
    float* W1   = alloc(IMG);    // xc / qtmp / combine tmp
    float* W2   = alloc(IMG);    // high / agg
    float* W3   = alloc(IMG);    // low
    float* W4   = alloc(IMG);    // q (after dwconv)
    float* W5   = alloc(IMG2);   // kv 1x1 out
    float* W6   = alloc(IMG2);   // kv after dwconv
    float* W7   = alloc(IMG);    // high2
    float* W8   = alloc(IMG);    // low2
    float* MXMN = alloc((size_t)BATCH * 2 * NP);
    float* SW   = alloc((size_t)BATCH * NP);
    float* POOL = alloc(512);
    float* FCO  = alloc(4096);
    int*   THR  = (int*)alloc(64);
    float* QSS  = alloc(512);
    float* KSS  = alloc(512);
    float* ATTN = alloc(4096);
    float* WEFF = alloc((size_t)BATCH * 64 * 64);
    float* LAVG = alloc(512);
    float* LMAX = alloc(512);
    float* CWB  = alloc(512);

    const dim3 gP64(NP / 256, 64, BATCH);
    const dim3 gP128(NP / 256, 128, BATCH);
    const dim3 gP1(NP / 256, 1, BATCH);
    const dim3 gG64(NP / 256, 4, BATCH);    // gemm Cout=64 (2 px-tiles/wave)
    const dim3 gG128(NP / 256, 8, BATCH);   // gemm Cout=128

    // channel-cross-attention: q from xin, kv from yin, result -> outbuf
    auto run_cca = [&](const float* xin, const float* yin, int idx, float* outbuf,
                       bool finalEpi) {
        const float* qw   = cca_qw   + (size_t)idx * 64 * 64;
        const float* qdw  = cca_qdw  + (size_t)idx * 64 * 9;
        const float* kvw  = cca_kvw  + (size_t)idx * 128 * 64;
        const float* kvdw = cca_kvdw + (size_t)idx * 128 * 9;
        const float* prw  = cca_projw+ (size_t)idx * 64 * 64;
        const float* tmp  = cca_temp + (size_t)idx * 8;
        gemm64_wmma<false, false, false><<<gG64, 256, 0, stream>>>(
            xin, 64, qw, W1, 64, NP, nullptr, nullptr, nullptr, nullptr);
        dwconv3x3_k<<<gP64, 256, 0, stream>>>(W1, qdw, W4, 64);
        gemm64_wmma<false, false, false><<<gG128, 256, 0, stream>>>(
            yin, 64, kvw, W5, 128, NP, nullptr, nullptr, nullptr, nullptr);
        dwconv3x3_k<<<gP128, 256, 0, stream>>>(W5, kvdw, W6, 128);
        sumsq_k<<<512, 256, 0, stream>>>(W4, QSS, 64, NP);
        sumsq_k<<<512, 256, 0, stream>>>(W6, KSS, 128, NP);      // k = kv[:, :64]
        gram_k<<<64, 256, 0, stream>>>(W4, W6, QSS, KSS, tmp, ATTN, NP);
        weff_k<<<8, 256, 0, stream>>>(ATTN, prw, WEFF);
        // v = kv[:, 64:128]; batched (per-b) 64x64 GEMM with fused proj
        if (finalEpi) {
            gemm64_wmma<true, false, true><<<gG64, 256, 0, stream>>>(
                W6 + (size_t)64 * NP, 128, WEFF, outbuf, 64, NP,
                nullptr, para1, y, para2);
        } else {
            gemm64_wmma<true, false, false><<<gG64, 256, 0, stream>>>(
                W6 + (size_t)64 * NP, 128, WEFF, outbuf, 64, NP,
                nullptr, nullptr, nullptr, nullptr);
        }
    };

    // 1) resize + first conv
    resize_k<<<(BATCH * 3 * NP) / 256, 256, 0, stream>>>(x, XR);
    conv3x3_k<<<gP64, 256, 0, stream>>>(XR, conv1_w, W1);        // xc -> W1

    // 2) pooled MLP thresholds (each in {0,1})
    mean_k<<<512, 256, 0, stream>>>(W1, POOL, 64, NP);
    thr_k<<<1, 8, 0, stream>>>(POOL, rate_w1, rate_w2, THR);

    // 3) exact low/high split via 4-coefficient DFT
    dft4_k<<<512, 256, 0, stream>>>(W1, FCO);
    lowhigh_k<<<gP64, 256, 0, stream>>>(W1, FCO, THR, W2 /*high*/, W3 /*low*/);

    // 4) CCA(high, y) -> high2 (W7); CCA(low, y) -> low2 (W8)
    run_cca(W2, y, 0, W7, false);
    run_cca(W3, y, 1, W8, false);

    // 5) fre_refine(low2=W8, high2=W7)
    chan_maxmean_k<<<gP1, 256, 0, stream>>>(W7, MXMN);
    conv7x7_sig_k<<<gP1, 256, 0, stream>>>(MXMN, spat_w, SW);
    avgmax_k<<<512, 256, 0, stream>>>(W8, LAVG, LMAX, 64, NP);
    cw_k<<<8, 64, 0, stream>>>(LAVG, LMAX, cg_w1, cg_w2, CWB);
    combine_k<<<gP64, 256, 0, stream>>>(W8, W7, SW, CWB, W1);
    gemm64_wmma<false, true, false><<<gG64, 256, 0, stream>>>(
        W1, 64, rproj_w, W2 /*agg*/, 64, NP, rproj_b, nullptr, nullptr, nullptr);

    // 6) final CCA(y, agg) with fused epilogue: out = cca*para1 + y*para2
    run_cca(y, W2, 2, out, true);
}
